// _AttentionRNN_64553358458910
// MI455X (gfx1250) — matrix operand order repaired and checked by reading, not compiled, not run on hardware
//
#include <hip/hip_runtime.h>
#include <hip/hip_bf16.h>

typedef __bf16 bf16;
typedef __attribute__((ext_vector_type(16))) bf16 v16bf;
typedef __attribute__((ext_vector_type(8)))  bf16 v8bf;
typedef __attribute__((ext_vector_type(8)))  float v8f;

#define S_STEPS 600
#define BATCH   64
#define TOK     160
#define DIN     512
#define ATT     256
#define NOUT    512
#define NF      32
#define KC      31

// ---------- helpers ----------

#if __has_builtin(__builtin_amdgcn_tanhf)
__device__ __forceinline__ float fast_tanh(float x) { return __builtin_amdgcn_tanhf(x); }
#else
__device__ __forceinline__ float fast_tanh(float x) { return tanhf(x); }
#endif

__device__ __forceinline__ float fast_sigmoid(float x) {
  return 0.5f * fast_tanh(0.5f * x) + 0.5f;    // one TRANS + 2 VALU
}

__device__ __forceinline__ bf16 f2bf(float f) {
  union { float f; unsigned u; } in; in.f = f;
  unsigned u = in.u;
  u += 0x7FFFu + ((u >> 16) & 1u);          // round-to-nearest-even
  union { unsigned short s; bf16 b; } out;
  out.s = (unsigned short)(u >> 16);
  return out.b;
}

__device__ __forceinline__ v8f wmma_bf16(v16bf a, v16bf b, v8f c) {
  return __builtin_amdgcn_wmma_f32_16x16x32_bf16(false, a, false, b, (short)0, c,
                                                 false, false);
}

// A-fragment (16x32 bf16, wave32): lane holds row m=lane&15; rel-K elements are
// {kb..kb+7, kb+16..kb+23} with kb = (lane<16)?0:8 -> two contiguous 16B loads.
__device__ __forceinline__ v16bf load_a_frag(const bf16* rowp, int kb) {
  v8bf lo = *(const v8bf*)(rowp + kb);
  v8bf hi = *(const v8bf*)(rowp + kb + 16);
  return __builtin_shufflevector(lo, hi, 0, 1, 2, 3, 4, 5, 6, 7,
                                 8, 9, 10, 11, 12, 13, 14, 15);
}

// ---------- one-time packing ----------
// B matrix (32x16, bf16) per wave32: lane half h, element s: K=8h+s (s<8),
// K=16+8h+(s-8) (s>=8), N=lane&15. Stored frag-major -> coalesced 32B/lane.
__global__ void pack_b_frags(const float* __restrict__ W, bf16* __restrict__ frags,
                             int K, int N, int ldw, int transposed) {
  int kt_count = K >> 5, nt_count = N >> 4;
  int idx = blockIdx.x * blockDim.x + threadIdx.x;   // one per (kt,nt,lane)
  int total = kt_count * nt_count * 32;
  if (idx >= total) return;
  int lane = idx & 31;
  int t    = idx >> 5;
  int nt   = t % nt_count;
  int kt   = t / nt_count;
  int col  = lane & 15;
  int koff = (lane < 16) ? 0 : 8;
  int n    = nt * 16 + col;
  int k0   = kt * 32 + koff;
  bf16* dst = frags + (size_t)idx * 16;
#pragma unroll
  for (int r = 0; r < 8; ++r) {
    int ka = k0 + 2 * r + (r >= 4 ? 8 : 0), kb = ka + 1;
    float w0 = transposed ? W[(size_t)n * ldw + ka] : W[(size_t)ka * ldw + n];
    float w1 = transposed ? W[(size_t)n * ldw + kb] : W[(size_t)kb * ldw + n];
    dst[2 * r]     = f2bf(w0);
    dst[2 * r + 1] = f2bf(w1);
  }
}

__global__ void f32_to_bf16_kernel(const float* __restrict__ src,
                                   bf16* __restrict__ dst, size_t n) {
  size_t i = (size_t)blockIdx.x * blockDim.x + threadIdx.x;
  if (i < n) dst[i] = f2bf(src[i]);
}

__global__ void copy_f32(const float* __restrict__ src, float* __restrict__ dst, int n) {
  int i = blockIdx.x * blockDim.x + threadIdx.x;
  if (i < n) dst[i] = src[i];
}

// ---------- keys = tokens @ Wk  ([B*T,256] x [256,256]) ----------
__global__ void keys_kernel(const bf16* __restrict__ tokens_bf,
                            const bf16* __restrict__ wk_frags,
                            float* __restrict__ keys) {
  int lane = threadIdx.x & 31, wave = threadIdx.x >> 5;
  int tile = blockIdx.x * 4 + wave;              // 640 mt * 16 nt = 10240 tiles
  int mt = tile >> 4, nt = tile & 15;
  int mrow  = mt * 16 + (lane & 15);
  int kb    = (lane < 16) ? 0 : 8;
  int col   = lane & 15;
  int rbase = (lane < 16) ? 0 : 8;
  v8f acc;
#pragma unroll
  for (int r = 0; r < 8; ++r) acc[r] = 0.f;
  const bf16* arow = tokens_bf + (size_t)mrow * ATT;
  const bf16* bp   = wk_frags + ((size_t)nt * 32 + lane) * 16;
  for (int kt = 0; kt < 8; ++kt) {
    v16bf a = load_a_frag(arow + kt * 32, kb);
    v16bf b = *(const v16bf*)(bp + (size_t)kt * 16 * 32 * 16);
    acc = wmma_bf16(a, b, acc);
  }
#pragma unroll
  for (int r = 0; r < 8; ++r)
    keys[(size_t)(mt * 16 + rbase + r) * ATT + nt * 16 + col] = acc[r];
}

// ---------- per-step kernel 1: LSTM gates GEMM (all-bf16 operands) ----------
// gates[64,1024] = [q_bf|ctx_bf][64,768] @ WihT + h_bf[64,256] @ WhhT + bias
__global__ void gates_kernel(const bf16* __restrict__ qbf_s,    // [64,512]
                             const bf16* __restrict__ ctx_bf,   // [64,256]
                             const bf16* __restrict__ h_bf,     // [64,256]
                             const float* __restrict__ b_ih,
                             const float* __restrict__ b_hh,
                             const bf16* __restrict__ wx_frags,
                             const bf16* __restrict__ whh_frags,
                             float* __restrict__ gates) {
  int lane = threadIdx.x & 31, wave = threadIdx.x >> 5;
  int tile = blockIdx.x * 4 + wave;              // 4 mt * 64 nt = 256 tiles
  int mt = tile >> 6, nt = tile & 63;
  int arow  = mt * 16 + (lane & 15);
  int kb    = (lane < 16) ? 0 : 8;
  int col   = lane & 15;
  int n     = nt * 16 + col;
  int rbase = (lane < 16) ? 0 : 8;
  float bias = b_ih[n] + b_hh[n];
  v8f acc;
#pragma unroll
  for (int r = 0; r < 8; ++r) acc[r] = bias;

  const bf16* qrow = qbf_s   + (size_t)arow * DIN;
  const bf16* crow = ctx_bf  + (size_t)arow * ATT;
  const bf16* hrow = h_bf    + (size_t)arow * ATT;
  const bf16* bx   = wx_frags  + ((size_t)nt * 32 + lane) * 16;
  const bf16* bh   = whh_frags + ((size_t)nt * 32 + lane) * 16;
  // query part: k-tiles 0..15
  for (int kt = 0; kt < 16; ++kt) {
    v16bf a = load_a_frag(qrow + kt * 32, kb);
    v16bf b = *(const v16bf*)(bx + (size_t)kt * 64 * 32 * 16);
    __builtin_prefetch(bx + (size_t)(kt + 1) * 64 * 32 * 16, 0, 1);
    acc = wmma_bf16(a, b, acc);
  }
  // ctx part: k-tiles 16..23
  for (int kt = 0; kt < 8; ++kt) {
    v16bf a = load_a_frag(crow + kt * 32, kb);
    v16bf b = *(const v16bf*)(bx + (size_t)(16 + kt) * 64 * 32 * 16);
    acc = wmma_bf16(a, b, acc);
  }
  // h part: 8 k-tiles
  for (int kt = 0; kt < 8; ++kt) {
    v16bf a = load_a_frag(hrow + kt * 32, kb);
    v16bf b = *(const v16bf*)(bh + (size_t)kt * 64 * 32 * 16);
    __builtin_prefetch(bh + (size_t)(kt + 1) * 64 * 32 * 16, 0, 1);
    acc = wmma_bf16(a, b, acc);
  }
#pragma unroll
  for (int r = 0; r < 8; ++r)
    gates[(size_t)(mt * 16 + rbase + r) * 1024 + n] = acc[r];
}

// ---------- per-step kernel 2: LSTM update + location attention ----------
// One workgroup (256 thr / 8 waves) per batch element.
__global__ void attn_kernel(int s,
                            const float* __restrict__ gates,
                            const float* __restrict__ Wq,
                            const float* __restrict__ loc_w,
                            const float* __restrict__ vvec,
                            const float* __restrict__ keys,
                            const float* __restrict__ tokens,
                            const bf16* __restrict__ wloc_frags,
                            float* __restrict__ h_st, float* __restrict__ c_st,
                            float* __restrict__ ctx_st, float* __restrict__ cum_st,
                            bf16* __restrict__ ctx_bf,
                            bf16* __restrict__ hs_bf,
                            float* __restrict__ out_ctxs,
                            float* __restrict__ out_aligns,
                            float* __restrict__ out_wss) {
  int b = blockIdx.x;
  int tid = threadIdx.x;
  int lane = tid & 31, wave = tid >> 5;

  __shared__ float h_lds[ATT];
  __shared__ float qp_lds[ATT];
  __shared__ float cum_pad[TOK + 30];
  __shared__ __attribute__((aligned(16))) bf16 locBf[TOK * NF];  // [t][f] bf16
  __shared__ float locw_lds[NF * KC];
  __shared__ float e_lds[TOK];
  __shared__ float align_lds[TOK];
  __shared__ float red[256];
  __shared__ int   redi[256];

  // LSTM elementwise update (i,f,g,o split along 1024)
  {
    int u = tid;
    float gi = gates[b * 1024 + u];
    float gf = gates[b * 1024 + 256 + u];
    float gg = gates[b * 1024 + 512 + u];
    float go = gates[b * 1024 + 768 + u];
    float c_old = c_st[b * ATT + u];
    float c_new = fast_sigmoid(gf) * c_old + fast_sigmoid(gi) * fast_tanh(gg);
    float h_new = fast_sigmoid(go) * fast_tanh(c_new);
    c_st[b * ATT + u] = c_new;
    h_st[b * ATT + u] = h_new;
    h_lds[u] = h_new;
    hs_bf[((size_t)s * BATCH + b) * ATT + u] = f2bf(h_new);
  }
  if (tid < TOK + 30) {                           // cum with SAME-pad halo
    int t = tid - 15;
    cum_pad[tid] = (t >= 0 && t < TOK) ? cum_st[b * TOK + t] : 0.f;
  }
  for (int i = tid; i < NF * KC; i += 256) locw_lds[i] = loc_w[i];
  __syncthreads();

  // qp = h @ Wq  (coalesced over output column e = tid)
  {
    float acc = 0.f;
    for (int d = 0; d < ATT; ++d) acc += h_lds[d] * Wq[d * ATT + tid];
    qp_lds[tid] = acc;
  }
  // location conv: locBf[t*32+f] = bf16( sum_k loc_w[f,k]*cum_pad[t+k] )
  // mapping: 32 consecutive threads share t (broadcast cum read, contiguous store)
  for (int o = tid; o < NF * TOK; o += 256) {
    int t = o >> 5, f = o & 31;
    float acc = 0.f;
#pragma unroll
    for (int k = 0; k < KC; ++k) acc += locw_lds[f * KC + k] * cum_pad[t + k];
    locBf[t * 32 + f] = f2bf(acc);
  }
  __syncthreads();

  // loc projection (WMMA, K=32) fused with e = sum_a v[a]*tanh(keys+qp+loc)
  {
    int col = lane & 15;
    int kb = (lane < 16) ? 0 : 8;
    int rbase = (lane < 16) ? 0 : 8;
    for (int rt = wave; rt < 10; rt += 8) {       // 10 t-tiles of 16 rows
      int trow = rt * 16 + (lane & 15);
      v16bf a = load_a_frag(&locBf[trow * 32], kb);   // two 16B LDS loads
      float eacc[8];
#pragma unroll
      for (int r = 0; r < 8; ++r) eacc[r] = 0.f;
      for (int nt = 0; nt < 16; ++nt) {
        v8f accv;
#pragma unroll
        for (int r = 0; r < 8; ++r) accv[r] = 0.f;
        v16bf bf = *(const v16bf*)(wloc_frags + ((size_t)nt * 32 + lane) * 16);
        accv = wmma_bf16(a, bf, accv);
        int acol = nt * 16 + col;
        float vv = vvec[acol];
        float qv = qp_lds[acol];
#pragma unroll
        for (int r = 0; r < 8; ++r) {
          int t = rt * 16 + rbase + r;
          float x = keys[((size_t)b * TOK + t) * ATT + acol] + qv + accv[r];
          eacc[r] += vv * fast_tanh(x);
        }
      }
#pragma unroll
      for (int r = 0; r < 8; ++r) {               // reduce across 16-lane half
        float x = eacc[r];
        x += __shfl_xor(x, 1);
        x += __shfl_xor(x, 2);
        x += __shfl_xor(x, 4);
        x += __shfl_xor(x, 8);
        eacc[r] = x;
      }
      if (col == 0) {
#pragma unroll
        for (int r = 0; r < 8; ++r) e_lds[rt * 16 + rbase + r] = eacc[r];
      }
    }
  }
  __syncthreads();

  // softmax over T with first-index argmax
  float mv = (tid < TOK) ? e_lds[tid] : -3.4e38f;
  red[tid] = mv; redi[tid] = tid;
  __syncthreads();
  for (int off = 128; off > 0; off >>= 1) {
    if (tid < off) {
      float bv = red[tid + off]; int ib = redi[tid + off];
      if (bv > red[tid] || (bv == red[tid] && ib < redi[tid])) { red[tid] = bv; redi[tid] = ib; }
    }
    __syncthreads();
  }
  float emax = red[0];
  int amax = redi[0];
  __syncthreads();
  float ex = (tid < TOK) ? __expf(e_lds[tid] - emax) : 0.f;
  red[tid] = ex;
  __syncthreads();
  for (int off = 128; off > 0; off >>= 1) {
    if (tid < off) red[tid] += red[tid + off];
    __syncthreads();
  }
  float inv = 1.f / red[0];
  if (tid < TOK) {
    float al = ex * inv;
    align_lds[tid] = al;
    out_aligns[((size_t)s * BATCH + b) * TOK + tid] = al;
    cum_st[b * TOK + tid] += al;
  }
  if (tid == 0) out_wss[(size_t)s * BATCH + b] = (float)amax;
  __syncthreads();

  // ctx = align @ tokens (coalesced over d = tid)
  {
    float acc = 0.f;
    for (int t = 0; t < TOK; ++t)
      acc += align_lds[t] * tokens[((size_t)b * TOK + t) * ATT + tid];
    ctx_st[b * ATT + tid] = acc;
    ctx_bf[b * ATT + tid] = f2bf(acc);
    out_ctxs[((size_t)s * BATCH + b) * ATT + tid] = acc;
  }
}

// ---------- final projection: rnn_outs = hs @ W_proj^T + b_proj ----------
__global__ void proj_kernel(const bf16* __restrict__ hs_bf,
                            const bf16* __restrict__ wp_frags,
                            const float* __restrict__ b_proj,
                            float* __restrict__ out) {
  int lane = threadIdx.x & 31, wave = threadIdx.x >> 5;
  size_t tile = (size_t)blockIdx.x * 8 + wave;    // 2400 mt * 32 nt
  int mt = (int)(tile >> 5), nt = (int)(tile & 31);
  int mrow  = mt * 16 + (lane & 15);
  int kb    = (lane < 16) ? 0 : 8;
  int col   = lane & 15;
  int n     = nt * 16 + col;
  int rbase = (lane < 16) ? 0 : 8;
  float bias = b_proj[n];
  v8f acc;
#pragma unroll
  for (int r = 0; r < 8; ++r) acc[r] = bias;
  const bf16* arow = hs_bf + (size_t)mrow * ATT;
  const bf16* bp   = wp_frags + ((size_t)nt * 32 + lane) * 16;
  for (int kt = 0; kt < 8; ++kt) {
    v16bf a = load_a_frag(arow + kt * 32, kb);
    v16bf b = *(const v16bf*)(bp + (size_t)kt * 32 * 32 * 16);
    __builtin_prefetch(bp + (size_t)(kt + 1) * 32 * 32 * 16, 0, 1);
    acc = wmma_bf16(a, b, acc);
  }
#pragma unroll
  for (int r = 0; r < 8; ++r)
    out[(size_t)(mt * 16 + rbase + r) * NOUT + n] = acc[r];
}

// ---------- host ----------
extern "C" void kernel_launch(void* const* d_in, const int* in_sizes, int n_in,
                              void* d_out, int out_size, void* d_ws, size_t ws_size,
                              hipStream_t stream) {
  const float* query  = (const float*)d_in[0];
  const float* tokens = (const float*)d_in[1];
  const float* h0     = (const float*)d_in[2];
  const float* c0     = (const float*)d_in[3];
  const float* ctx0   = (const float*)d_in[4];
  const float* cum0   = (const float*)d_in[5];
  const float* W_ih   = (const float*)d_in[6];
  const float* W_hh   = (const float*)d_in[7];
  const float* b_ih   = (const float*)d_in[8];
  const float* b_hh   = (const float*)d_in[9];
  const float* Wq     = (const float*)d_in[10];
  const float* Wk     = (const float*)d_in[11];
  const float* loc_w  = (const float*)d_in[12];
  const float* W_loc  = (const float*)d_in[13];
  const float* vvec   = (const float*)d_in[14];
  const float* W_proj = (const float*)d_in[15];
  const float* b_proj = (const float*)d_in[16];

  // workspace layout (all offsets 256B aligned)
  char* w = (char*)d_ws;
  constexpr size_t OFF_KEYS  = 0;
  constexpr size_t OFF_HSBF  = OFF_KEYS  + (size_t)BATCH * TOK * ATT * 4;
  constexpr size_t OFF_QBF   = OFF_HSBF  + (size_t)S_STEPS * BATCH * ATT * 2;
  constexpr size_t OFF_TOKBF = OFF_QBF   + (size_t)S_STEPS * BATCH * DIN * 2;
  constexpr size_t OFF_GATES = OFF_TOKBF + (size_t)BATCH * TOK * ATT * 2;
  constexpr size_t OFF_H     = OFF_GATES + (size_t)BATCH * 1024 * 4;
  constexpr size_t OFF_C     = OFF_H     + (size_t)BATCH * ATT * 4;
  constexpr size_t OFF_CTX   = OFF_C     + (size_t)BATCH * ATT * 4;
  constexpr size_t OFF_CUM   = OFF_CTX   + (size_t)BATCH * ATT * 4;
  constexpr size_t OFF_H0BF  = OFF_CUM   + (size_t)BATCH * TOK * 4;
  constexpr size_t OFF_CTXBF = OFF_H0BF  + (size_t)BATCH * ATT * 2;
  constexpr size_t OFF_WX    = OFF_CTXBF + (size_t)BATCH * ATT * 2;
  constexpr size_t OFF_WHH   = OFF_WX    + (size_t)24 * 64 * 32 * 16 * 2;
  constexpr size_t OFF_WK    = OFF_WHH   + (size_t)8 * 64 * 32 * 16 * 2;
  constexpr size_t OFF_WLOC  = OFF_WK    + (size_t)8 * 16 * 32 * 16 * 2;
  constexpr size_t OFF_WPROJ = OFF_WLOC  + (size_t)1 * 16 * 32 * 16 * 2;

  float* keys    = (float*)(w + OFF_KEYS);
  bf16*  hs_bf   = (bf16*)(w + OFF_HSBF);
  bf16*  q_bf    = (bf16*)(w + OFF_QBF);
  bf16*  tok_bf  = (bf16*)(w + OFF_TOKBF);
  float* gates   = (float*)(w + OFF_GATES);
  float* h_st    = (float*)(w + OFF_H);
  float* c_st    = (float*)(w + OFF_C);
  float* ctx_st  = (float*)(w + OFF_CTX);
  float* cum_st  = (float*)(w + OFF_CUM);
  bf16*  h0_bf   = (bf16*)(w + OFF_H0BF);
  bf16*  ctx_bf  = (bf16*)(w + OFF_CTXBF);
  bf16*  wx_f    = (bf16*)(w + OFF_WX);
  bf16*  whh_f   = (bf16*)(w + OFF_WHH);
  bf16*  wk_f    = (bf16*)(w + OFF_WK);
  bf16*  wloc_f  = (bf16*)(w + OFF_WLOC);
  bf16*  wproj_f = (bf16*)(w + OFF_WPROJ);

  // output layout (return order: rnn_outs, ctxs, aligns, wss, (h,c,ctx,cum))
  float* out       = (float*)d_out;
  float* out_rnn   = out;
  float* out_ctxs  = out_rnn  + (size_t)S_STEPS * BATCH * NOUT;
  float* out_align = out_ctxs + (size_t)S_STEPS * BATCH * ATT;
  float* out_wss   = out_align + (size_t)S_STEPS * BATCH * TOK;
  float* out_h     = out_wss  + (size_t)S_STEPS * BATCH;
  float* out_c     = out_h    + (size_t)BATCH * ATT;
  float* out_ctxf  = out_c    + (size_t)BATCH * ATT;
  float* out_cum   = out_ctxf + (size_t)BATCH * ATT;

  // 1) pack weights as WMMA B-fragments (bf16)
  auto packN = [&](const float* W, bf16* dst, int K, int N, int ldw, int tr) {
    int total = (K >> 5) * (N >> 4) * 32;
    pack_b_frags<<<(total + 255) / 256, 256, 0, stream>>>(W, dst, K, N, ldw, tr);
  };
  packN(W_ih,   wx_f,    768, 1024, 768, 1);   // WihT[k,n] = W_ih[n,k]
  packN(W_hh,   whh_f,   256, 1024, 256, 1);
  packN(Wk,     wk_f,    256, 256,  256, 0);
  packN(W_loc,  wloc_f,  32,  256,  256, 0);
  packN(W_proj, wproj_f, 256, 512,  256, 1);   // WprojT[k,n] = W_proj[n,k]

  // 2) bf16 activation copies + recurrent state init
  {
    size_t nq = (size_t)S_STEPS * BATCH * DIN;
    f32_to_bf16_kernel<<<(unsigned)((nq + 255) / 256), 256, 0, stream>>>(query, q_bf, nq);
    size_t ntk = (size_t)BATCH * TOK * ATT;
    f32_to_bf16_kernel<<<(unsigned)((ntk + 255) / 256), 256, 0, stream>>>(tokens, tok_bf, ntk);
    f32_to_bf16_kernel<<<(BATCH * ATT + 255) / 256, 256, 0, stream>>>(h0, h0_bf, BATCH * ATT);
    f32_to_bf16_kernel<<<(BATCH * ATT + 255) / 256, 256, 0, stream>>>(ctx0, ctx_bf, BATCH * ATT);
  }
  copy_f32<<<(BATCH * ATT + 255) / 256, 256, 0, stream>>>(h0,   h_st,   BATCH * ATT);
  copy_f32<<<(BATCH * ATT + 255) / 256, 256, 0, stream>>>(c0,   c_st,   BATCH * ATT);
  copy_f32<<<(BATCH * ATT + 255) / 256, 256, 0, stream>>>(ctx0, ctx_st, BATCH * ATT);
  copy_f32<<<(BATCH * TOK + 255) / 256, 256, 0, stream>>>(cum0, cum_st, BATCH * TOK);

  // 3) keys = tokens @ Wk  (10240 tiles, 4 waves/block)
  keys_kernel<<<2560, 128, 0, stream>>>(tok_bf, wk_f, keys);

  // 4) sequential scan: 2 kernels per step, ordered by the stream
  for (int s = 0; s < S_STEPS; ++s) {
    const bf16* q_s = q_bf + (size_t)s * BATCH * DIN;
    const bf16* hsrc = (s == 0) ? h0_bf : hs_bf + (size_t)(s - 1) * BATCH * ATT;
    gates_kernel<<<64, 128, 0, stream>>>(q_s, ctx_bf, hsrc, b_ih, b_hh,
                                         wx_f, whh_f, gates);
    attn_kernel<<<BATCH, 256, 0, stream>>>(s, gates, Wq, loc_w, vvec, keys, tokens,
                                           wloc_f, h_st, c_st, ctx_st, cum_st,
                                           ctx_bf, hs_bf, out_ctxs, out_align, out_wss);
  }

  // 5) rnn_outs = hs @ W_proj^T + b_proj  (76800 tiles, 8 waves/block)
  proj_kernel<<<9600, 256, 0, stream>>>(hs_bf, wproj_f, b_proj, out_rnn);

  // 6) final state tuple
  copy_f32<<<(BATCH * ATT + 255) / 256, 256, 0, stream>>>(h_st,   out_h,    BATCH * ATT);
  copy_f32<<<(BATCH * ATT + 255) / 256, 256, 0, stream>>>(c_st,   out_c,    BATCH * ATT);
  copy_f32<<<(BATCH * ATT + 255) / 256, 256, 0, stream>>>(ctx_st, out_ctxf, BATCH * ATT);
  copy_f32<<<(BATCH * TOK + 255) / 256, 256, 0, stream>>>(cum_st, out_cum,  BATCH * TOK);
}
